// Attention_LoRA_SRT_76141180223914
// MI455X (gfx1250) — compile-verified, matrix operand order checked by reading, not run on hardware
//
#include <hip/hip_runtime.h>

typedef unsigned short u16;
typedef __attribute__((ext_vector_type(16))) __bf16 v16bf;
typedef __attribute__((ext_vector_type(8)))  float  v8f;
typedef __attribute__((ext_vector_type(4)))  unsigned int v4u;
typedef __attribute__((ext_vector_type(8)))  int   v8i;
typedef __attribute__((ext_vector_type(4)))  int   v4i;

#define B_  32
#define N_  577
#define C_  768
#define H_  12
#define D_  64
#define M_  (B_ * N_)                    // 18464 tokens
#define QKV_STRIDE (B_ * H_ * N_ * D_)   // 14,180,352 elements per q/k/v plane

#if defined(__HIP_DEVICE_COMPILE__) && __has_builtin(__builtin_amdgcn_tensor_load_to_lds)
#define HAS_TDM 1
#else
#define HAS_TDM 0
#endif

union U4BF8 { uint4 v; u16 s[8]; };

static __device__ __forceinline__ u16 f2bf(float f) {
    unsigned u = __float_as_uint(f);
    u += 0x7FFFu + ((u >> 16) & 1u);   // round-to-nearest-even
    return (u16)(u >> 16);
}

// 16-lane xor "butterfly" exchange: v_permlane16_b32 (VALU) when available,
// so the softmax reductions stay off the LDS pipe (ds_bpermute fallback).
template <int MASK>
static __device__ __forceinline__ float xor16f(float x) {
#if defined(__gfx1250__) && __has_builtin(__builtin_amdgcn_permlane16)
    constexpr unsigned S0 = (MASK == 1) ? 0x67452301u
                          : (MASK == 2) ? 0x54761032u
                          : (MASK == 4) ? 0x32107654u
                                        : 0xFEDCBA98u;   // MASK == 8
    constexpr unsigned S1 = (MASK == 1) ? 0xEFCDAB89u
                          : (MASK == 2) ? 0xDCFE98BAu
                          : (MASK == 4) ? 0xBA98FEDCu
                                        : 0x76543210u;
    unsigned u = __float_as_uint(x);
    u = __builtin_amdgcn_permlane16(u, u, S0, S1, false, false);
    return __uint_as_float(u);
#else
    return __shfl_xor(x, MASK, 16);
#endif
}

// Build a 16x32 A-frag (or 32x16 B-frag) from a row of bf16 data.
// Elements 0..7  <- [g*8 .. g*8+7], elements 8..15 <- [16+g*8 .. 16+g*8+7].
static __device__ __forceinline__ v16bf make_frag(const u16* __restrict__ row, int g) {
    U4BF8 lo, hi;
    lo.v = *(const uint4*)(row + g * 8);
    hi.v = *(const uint4*)(row + 16 + g * 8);
    v16bf f;
#pragma unroll
    for (int i = 0; i < 8; ++i) {
        f[i]     = __builtin_bit_cast(__bf16, lo.s[i]);
        f[i + 8] = __builtin_bit_cast(__bf16, hi.s[i]);
    }
    return f;
}

static __device__ __forceinline__ v8f wmma_bf16(v16bf a, v16bf b, v8f c) {
    return __builtin_amdgcn_wmma_f32_16x16x32_bf16(false, a, false, b, (short)0, c, false, false);
}

#if HAS_TDM
// Issue a TDM DMA of a [128 rows x 32 bf16] tile (tile start = row0, col kt of a
// row-major [rows_total x K] bf16 tensor) into LDS with 16B padding every 64B,
// i.e. LDS row stride 40 bf16 — matching the fragment-read layout below.
static __device__ __forceinline__ void tdm_issue(const u16* lds_ptr, const u16* gbase,
                                                 int row0, int rows_total, int K, int kt) {
    unsigned long long ga = (unsigned long long)(const void*)(gbase + (size_t)row0 * K + kt);
    v4u g0;
    g0[0] = 1u;                                          // count=1, no gather
    g0[1] = (unsigned)(unsigned long long)(const void*)lds_ptr;  // LDS byte offset
    g0[2] = (unsigned)ga;                                // global_addr[31:0]
    g0[3] = (unsigned)(ga >> 32) | (2u << 30);           // global_addr[56:32], type=2
    unsigned td0 = (unsigned)(K - kt);
    unsigned td1 = (unsigned)(rows_total - row0);        // short rows zero-fill
    v8i g1;
    g1[0] = (int)((1u << 16) | (1u << 20) | (3u << 22) | (3u << 25)); // 2B elems; pad 4 dw / 16 dw
    g1[1] = (int)((td0 & 0xFFFFu) << 16);                // tensor_dim0 lo
    g1[2] = (int)((td0 >> 16) | ((td1 & 0xFFFFu) << 16));// tensor_dim0 hi | tensor_dim1 lo
    g1[3] = (int)((td1 >> 16) | (32u << 16));            // tensor_dim1 hi | tile_dim0=32
    g1[4] = 128;                                         // tile_dim1=128, tile_dim2=0
    g1[5] = K;                                           // tensor_dim0_stride (elements)
    g1[6] = 0;
    g1[7] = 0;
    v4i z4 = {0, 0, 0, 0};
#if __clang_major__ >= 23
    v8i z8 = {0, 0, 0, 0, 0, 0, 0, 0};
    __builtin_amdgcn_tensor_load_to_lds(g0, g1, z4, z4, z8, 0);
#else
    __builtin_amdgcn_tensor_load_to_lds(g0, g1, z4, z4, 0);
#endif
}
#endif

// ---------------- Kernel 1: fp32 -> bf16 conversion of x ----------------
__global__ __launch_bounds__(256) void cvt_x_kernel(const float4* __restrict__ x,
                                                    u16* __restrict__ xb, int n4) {
    int i = blockIdx.x * 256 + threadIdx.x;
    if (i >= n4) return;
    float4 v = x[i];
    uint2 p;
    p.x = (unsigned)f2bf(v.x) | ((unsigned)f2bf(v.y) << 16);
    p.y = (unsigned)f2bf(v.z) | ((unsigned)f2bf(v.w) << 16);
    *(uint2*)(xb + (size_t)i * 4) = p;
}

// ------------- Kernel 2: merge LoRA into weights, cast to bf16 -------------
__global__ __launch_bounds__(256) void prep_w_kernel(
    const float* __restrict__ qkv_w, const float* __restrict__ proj_w,
    const float* __restrict__ Ak, const float* __restrict__ Bk,
    const float* __restrict__ Av, const float* __restrict__ Bv,
    const int* __restrict__ task,
    u16* __restrict__ wall, u16* __restrict__ pw) {
    const int TOT1 = 3 * C_ * C_;
    int idx = blockIdx.x * 256 + threadIdx.x;
    if (idx < TOT1) {
        int i = idx / C_, j = idx % C_;
        float w = qkv_w[idx];
        if (i >= C_ && (*task) > -1) {   // task > -0.5 for integer task
            const float* Bm = (i < 2 * C_) ? Bk : Bv;
            const float* Am = (i < 2 * C_) ? Ak : Av;
            int ii = (i < 2 * C_) ? (i - C_) : (i - 2 * C_);
            float acc = 0.f;
#pragma unroll 8
            for (int r = 0; r < 64; ++r)
                acc = fmaf(Bm[ii * 64 + r], Am[r * C_ + j], acc);
            w += acc;
        }
        wall[idx] = f2bf(w);
    } else {
        int k = idx - TOT1;
        if (k < C_ * C_) pw[k] = f2bf(proj_w[k]);
    }
}

// ------------- Kernels 3/5: bf16 WMMA GEMM  C[M,N] = A[M,K] * B[N,K]^T -------------
// Staging: TDM tensor_load_to_lds, double-buffered (DMA of tile k+1 overlaps compute
// of tile k; wave 0 issues, s_wait_tensorcnt + barrier publishes). Manual fallback kept.
// MODE 0: scatter into q/k/v planes ([B,H,N,d], bf16, q pre-scaled by d^-0.5)
// MODE 1: f32 out + bias (output projection)
template <int MODE>
__global__ __launch_bounds__(256) void gemm_bf16_kernel(
    const u16* __restrict__ A, const u16* __restrict__ Bm,
    int M, int Nn, int K,
    u16* __restrict__ qkv_out, float* __restrict__ f32_out,
    const float* __restrict__ bias) {
    constexpr int LDT = 40;                 // row stride (bf16), 80B: 16B aligned
    __shared__ u16 Alds[2][128 * LDT];
    __shared__ u16 Blds[2][128 * LDT];

    const int tid = threadIdx.x;
    const int lane = tid & 31, wid = tid >> 5;
    const int m0 = blockIdx.x * 128, n0 = blockIdx.y * 128;
    const int wm = (wid & 3) * 32;          // wave M offset (4 waves down)
    const int wn = (wid >> 2) * 64;         // wave N offset (2 waves across)
    const int g = lane >> 4, r = lane & 15;

    v8f acc[2][4];
#pragma unroll
    for (int mi = 0; mi < 2; ++mi)
#pragma unroll
        for (int ni = 0; ni < 4; ++ni)
#pragma unroll
            for (int q = 0; q < 8; ++q) acc[mi][ni][q] = 0.f;

#if HAS_TDM
    if (wid == 0) {
        tdm_issue(&Alds[0][0], A, m0, M, K, 0);
        tdm_issue(&Blds[0][0], Bm, n0, Nn, K, 0);
    }
    int cur = 0;
    for (int kt = 0; kt < K; kt += 32, cur ^= 1) {
        if (wid == 0) __builtin_amdgcn_s_wait_tensorcnt(0);   // buf[cur] landed
        __syncthreads();                                      // publish to all waves
        if (wid == 0 && kt + 32 < K) {                        // prefetch next tile via DMA
            tdm_issue(&Alds[cur ^ 1][0], A, m0, M, K, kt + 32);
            tdm_issue(&Blds[cur ^ 1][0], Bm, n0, Nn, K, kt + 32);
        }
        const u16* Ab = &Alds[cur][0];
        const u16* Bb = &Blds[cur][0];
#else
    const int lr = tid >> 1;                // tile row this thread loads
    const int lh = (tid & 1) * 16;          // 16-bf16 half of the 32-wide slice
    int arow = m0 + lr; if (arow > M - 1) arow = M - 1;
    const int brow = n0 + lr;               // N is a multiple of 128
    for (int kt = 0; kt < K; kt += 32) {
        const uint4* ag = (const uint4*)(A + (size_t)arow * K + kt + lh);
        const uint4* bg = (const uint4*)(Bm + (size_t)brow * K + kt + lh);
        uint4 a0 = ag[0], a1 = ag[1];
        uint4 b0 = bg[0], b1 = bg[1];
        *(uint4*)(&Alds[0][0] + lr * LDT + lh)     = a0;
        *(uint4*)(&Alds[0][0] + lr * LDT + lh + 8) = a1;
        *(uint4*)(&Blds[0][0] + lr * LDT + lh)     = b0;
        *(uint4*)(&Blds[0][0] + lr * LDT + lh + 8) = b1;
        __syncthreads();
        const u16* Ab = &Alds[0][0];
        const u16* Bb = &Blds[0][0];
#endif
        v16bf af0 = make_frag(Ab + (wm + r) * LDT, g);
        v16bf af1 = make_frag(Ab + (wm + 16 + r) * LDT, g);
        v16bf bf0 = make_frag(Bb + (wn + r) * LDT, g);
        v16bf bf1 = make_frag(Bb + (wn + 16 + r) * LDT, g);
        v16bf bf2 = make_frag(Bb + (wn + 32 + r) * LDT, g);
        v16bf bf3 = make_frag(Bb + (wn + 48 + r) * LDT, g);

        acc[0][0] = wmma_bf16(af0, bf0, acc[0][0]);
        acc[0][1] = wmma_bf16(af0, bf1, acc[0][1]);
        acc[0][2] = wmma_bf16(af0, bf2, acc[0][2]);
        acc[0][3] = wmma_bf16(af0, bf3, acc[0][3]);
        acc[1][0] = wmma_bf16(af1, bf0, acc[1][0]);
        acc[1][1] = wmma_bf16(af1, bf1, acc[1][1]);
        acc[1][2] = wmma_bf16(af1, bf2, acc[1][2]);
        acc[1][3] = wmma_bf16(af1, bf3, acc[1][3]);
        __syncthreads();
    }

    const int cl = lane & 15;
#pragma unroll
    for (int mi = 0; mi < 2; ++mi) {
#pragma unroll
        for (int ni = 0; ni < 4; ++ni) {
            int col = n0 + wn + ni * 16 + cl;
            float bv = (MODE == 1) ? bias[col] : 0.f;
#pragma unroll
            for (int q = 0; q < 8; ++q) {
                int row = m0 + wm + mi * 16 + g * 8 + q;
                if (row >= M) continue;
                float v = acc[mi][ni][q];
                if (MODE == 0) {
                    int which = col / C_;
                    int rem = col - which * C_;
                    int h = rem >> 6, dd = rem & 63;
                    int b = row / N_, nn = row - b * N_;
                    if (which == 0) v *= 0.125f;        // d^-0.5
                    qkv_out[(size_t)which * QKV_STRIDE +
                            (((size_t)(b * H_ + h)) * N_ + nn) * D_ + dd] = f2bf(v);
                } else {
                    f32_out[(size_t)row * Nn + col] = v + bv;
                }
            }
        }
    }
}

// ------------- Kernel 4: flash attention (per (b,h), 128 q-rows / WG) -------------
__global__ __launch_bounds__(256) void attn_kernel(
    const u16* __restrict__ qb, const u16* __restrict__ kb,
    const u16* __restrict__ vb, u16* __restrict__ ao) {
    __shared__ u16 Klds[32 * 72];          // K tile, row stride 72 (144B)
    __shared__ u16 Vt[64 * 40];            // V tile transposed [d][kn], stride 40
    __shared__ u16 Plds[8 * 16 * 40];      // per-wave P staging (D-layout -> A-layout)

    const int tid = threadIdx.x, lane = tid & 31, wid = tid >> 5;
    const int g = lane >> 4, cl = lane & 15;
    const int bh = blockIdx.y;
    const int b = bh / H_, h = bh % H_;
    const int qr0 = blockIdx.x * 128 + wid * 16;
    const size_t base = (size_t)bh * N_ * D_;

    int qrow = qr0 + cl; if (qrow > N_ - 1) qrow = N_ - 1;
    const u16* qrp = qb + base + (size_t)qrow * D_;
    v16bf Aq0 = make_frag(qrp, g);          // d 0..31
    v16bf Aq1 = make_frag(qrp + 32, g);     // d 32..63

    v8f Of[4];
#pragma unroll
    for (int i = 0; i < 4; ++i)
#pragma unroll
        for (int q = 0; q < 8; ++q) Of[i][q] = 0.f;
    float mrun[8], lrun[8];
#pragma unroll
    for (int q = 0; q < 8; ++q) { mrun[q] = -3.0e38f; lrun[q] = 0.f; }

    u16* Pw = Plds + wid * 16 * 40;

    for (int k0 = 0; k0 < N_; k0 += 32) {
        __syncthreads();
        {   // cooperative stage: K tile row-major, V tile transposed
            int row = tid >> 3, seg = tid & 7;
            int kr = k0 + row; if (kr > N_ - 1) kr = N_ - 1;
            uint4 kd = *(const uint4*)(kb + base + (size_t)kr * D_ + seg * 8);
            *(uint4*)(Klds + row * 72 + seg * 8) = kd;
            U4BF8 vv; vv.v = *(const uint4*)(vb + base + (size_t)kr * D_ + seg * 8);
#pragma unroll
            for (int i = 0; i < 8; ++i) Vt[(seg * 8 + i) * 40 + row] = vv.s[i];
            if (k0 + 32 < N_) {             // global_prefetch of next tile
                int pr = k0 + 32 + row; if (pr > N_ - 1) pr = N_ - 1;
                __builtin_prefetch(kb + base + (size_t)pr * D_ + seg * 8, 0, 0);
                __builtin_prefetch(vb + base + (size_t)pr * D_ + seg * 8, 0, 0);
            }
        }
        __syncthreads();

        // S = Q K^T  (two 16x16 key blocks, contraction over d split 32+32)
        v8f S0, S1;
#pragma unroll
        for (int q = 0; q < 8; ++q) { S0[q] = 0.f; S1[q] = 0.f; }
        v16bf b00 = make_frag(Klds + cl * 72, g);
        v16bf b01 = make_frag(Klds + cl * 72 + 32, g);
        v16bf b10 = make_frag(Klds + (16 + cl) * 72, g);
        v16bf b11 = make_frag(Klds + (16 + cl) * 72 + 32, g);
        S0 = wmma_bf16(Aq0, b00, S0); S0 = wmma_bf16(Aq1, b01, S0);
        S1 = wmma_bf16(Aq0, b10, S1); S1 = wmma_bf16(Aq1, b11, S1);

        const bool inv0 = (k0 + cl)      >= N_;
        const bool inv1 = (k0 + 16 + cl) >= N_;
        float P0[8], P1[8], fr[8];
#pragma unroll
        for (int q = 0; q < 8; ++q) {
            float s0 = inv0 ? -1.0e30f : S0[q];
            float s1 = inv1 ? -1.0e30f : S1[q];
            float t = fmaxf(s0, s1);                  // row reduce across 16 lanes (VALU)
            t = fmaxf(t, xor16f<1>(t));
            t = fmaxf(t, xor16f<2>(t));
            t = fmaxf(t, xor16f<4>(t));
            t = fmaxf(t, xor16f<8>(t));
            float mn = fmaxf(mrun[q], t);
            float f  = __expf(mrun[q] - mn);
            float p0 = __expf(s0 - mn), p1 = __expf(s1 - mn);
            float ls = p0 + p1;
            ls += xor16f<1>(ls);
            ls += xor16f<2>(ls);
            ls += xor16f<4>(ls);
            ls += xor16f<8>(ls);
            lrun[q] = lrun[q] * f + ls;
            mrun[q] = mn;
            fr[q] = f; P0[q] = p0; P1[q] = p1;
        }
#pragma unroll
        for (int db = 0; db < 4; ++db)
#pragma unroll
            for (int q = 0; q < 8; ++q) Of[db][q] *= fr[q];

        // stage P through per-wave LDS: D-layout -> row-major 16x32
#pragma unroll
        for (int q = 0; q < 8; ++q) {
            int prow = g * 8 + q;
            Pw[prow * 40 + cl]      = f2bf(P0[q]);
            Pw[prow * 40 + 16 + cl] = f2bf(P1[q]);
        }
        v16bf Pa = make_frag(Pw + cl * 40, g);        // A-frag 16x32 over kn
#pragma unroll
        for (int db = 0; db < 4; ++db) {
            v16bf bv = make_frag(Vt + (db * 16 + cl) * 40, g);
            Of[db] = wmma_bf16(Pa, bv, Of[db]);
        }
    }

    // normalize + store bf16 into ao [B*N, C]
#pragma unroll
    for (int q = 0; q < 8; ++q) {
        int n = qr0 + g * 8 + q;
        if (n >= N_) continue;
        float inv = 1.f / lrun[q];
#pragma unroll
        for (int db = 0; db < 4; ++db) {
            ao[((size_t)(b * N_ + n)) * C_ + h * 64 + db * 16 + cl] =
                f2bf(Of[db][q] * inv);
        }
    }
}

// ---------------- host-side launch ----------------
extern "C" void kernel_launch(void* const* d_in, const int* in_sizes, int n_in,
                              void* d_out, int out_size, void* d_ws, size_t ws_size,
                              hipStream_t stream) {
    const float* x      = (const float*)d_in[0];
    const int*   task   = (const int*)d_in[1];
    const float* qkv_w  = (const float*)d_in[2];
    const float* proj_w = (const float*)d_in[3];
    const float* proj_b = (const float*)d_in[4];
    const float* Ak     = (const float*)d_in[5];
    const float* Bk     = (const float*)d_in[6];
    const float* Av     = (const float*)d_in[7];
    const float* Bv     = (const float*)d_in[8];
    float* out = (float*)d_out;

    // workspace layout (bytes), ~118 MB total; x_bf16 region reused for attn output
    char* ws = (char*)d_ws;
    const size_t SZ_TOK = (size_t)M_ * C_ * 2;            // 28,360,704
    u16* xbf  = (u16*)(ws);                                // [M,C] bf16 (later: attn out)
    u16* wall = (u16*)(ws + SZ_TOK);                       // [3C,C] bf16 merged weights
    u16* pw   = (u16*)(ws + SZ_TOK + (size_t)3 * C_ * C_ * 2);
    u16* qkv  = (u16*)(ws + SZ_TOK + (size_t)4 * C_ * C_ * 2);  // 3 planes [B,H,N,d]
    u16* qb = qkv;
    u16* kb = qkv + (size_t)QKV_STRIDE;
    u16* vb = qkv + (size_t)2 * QKV_STRIDE;

    // 1) x -> bf16
    cvt_x_kernel<<<(M_ * C_ / 4 + 255) / 256, 256, 0, stream>>>(
        (const float4*)x, xbf, M_ * C_ / 4);

    // 2) merge LoRA into k/v weights, convert all weights to bf16
    prep_w_kernel<<<(3 * C_ * C_ + C_ * C_) / 256, 256, 0, stream>>>(
        qkv_w, proj_w, Ak, Bk, Av, Bv, task, wall, pw);

    // 3) fused qkv projection (q scaled by d^-0.5), scatter to [B,H,N,d] bf16
    gemm_bf16_kernel<0><<<dim3((M_ + 127) / 128, (3 * C_) / 128), 256, 0, stream>>>(
        xbf, wall, M_, 3 * C_, C_, qkv, nullptr, nullptr);

    // 4) flash attention, bf16 out into xbf (reused) as [B*N, C]
    attn_kernel<<<dim3((N_ + 127) / 128, B_ * H_), 256, 0, stream>>>(qb, kb, vb, xbf);

    // 5) output projection + bias, f32 out
    gemm_bf16_kernel<1><<<dim3((M_ + 127) / 128, C_ / 128), 256, 0, stream>>>(
        xbf, pw, M_, C_, C_, nullptr, out, proj_b);
}
